// QLinearLayer_46359876993657
// MI455X (gfx1250) — compile-verified, matrix operand order checked by reading, not run on hardware
//
#include <hip/hip_runtime.h>

typedef __attribute__((ext_vector_type(16))) _Float16 v16h;
typedef __attribute__((ext_vector_type(8)))  float    v8f;

#define BM   128
#define BN   128
#define BK   64            // K-depth per LDS stage (2 WMMA K-steps)
#define LDTK 68            // padded LDS row stride (halfs) to spread banks
#define CIN  1024
#define SBLK 16            // NVFP4 micro-block size along K
#define NSB  (CIN / SBLK)
#define NSTG (CIN / BK)    // 16 stages

// Gather one 16x16x32 f16 fragment (A or B) from an LDS row-major tile,
// following the CDNA5 wave32 VGPR layout (ISA 7.12.2):
//   lanes 0-15 : K dwords {0,2,4,6,16,18,20,22}
//   lanes 16-31: same + 8 halfs (K = 8..15, 24..31)
__device__ __forceinline__ v16h frag_from_lds(const _Float16* row, int hiHalf) {
  union { v16h h; int i[8]; } u;
#pragma unroll
  for (int e = 0; e < 8; ++e) {
    const int kd = ((e < 4) ? (2 * e) : (2 * e + 8)) + hiHalf;
    u.i[e] = *(const int*)(row + kd);
  }
  return u.h;
}

__global__ __launch_bounds__(256) void nvfp4_qlinear_wmma(
    const float* __restrict__ qx, const float* __restrict__ W,
    const float* __restrict__ sx, const float* __restrict__ sw,
    const float* __restrict__ gscale, const float* __restrict__ bias,
    float* __restrict__ out, int T, int Cout)
{
  __shared__ _Float16 As[2][BM * LDTK];   // double-buffered 128x64 f16 tiles
  __shared__ _Float16 Bs[2][BN * LDTK];

  const int tid     = threadIdx.x;
  const int lane    = tid & 31;
  const int wid     = tid >> 5;            // 0..7
  const int mw      = wid >> 1;            // 0..3 : 32-row band of the wave
  const int nw      = wid & 1;             // 0..1 : 64-col band of the wave
  const int laneRow = lane & 15;
  const int hiHalf  = (lane >> 4) << 3;    // 0 (lanes 0-15) or 8 (lanes 16-31)

  const int m0 = blockIdx.y * BM;
  const int n0 = blockIdx.x * BN;

  // Staging: 2 threads per row, 32 contiguous f32 (128B = two scale blocks).
  const int srow  = tid >> 1;              // 0..127 (row for both A and B tiles)
  const int shalf = (tid & 1) * 32;        // 0 or 32

  // Branch-free clamped source rows (epilogue stores are guarded instead).
  const int ta = min(m0 + srow, T - 1);
  const int ob = min(n0 + srow, Cout - 1);

  const float* aptr = qx + (size_t)ta * CIN + shalf;
  const float* sxa  = sx + (size_t)ta * NSB + (shalf / SBLK);
  const float* bptr = W  + (size_t)ob * CIN + shalf;
  const float* swb  = sw + (size_t)ob * NSB + (shalf / SBLK);

  _Float16* const dA0 = &As[0][srow * LDTK + shalf];
  _Float16* const dB0 = &Bs[0][srow * LDTK + shalf];
  const int bufStride = BM * LDTK;         // halfs between buffer 0 and buffer 1

  v8f acc[2][4];
#pragma unroll
  for (int i = 0; i < 2; ++i)
#pragma unroll
    for (int j = 0; j < 4; ++j) acc[i][j] = (v8f){};

  float ra[32], rb[32], sa0, sa1, sb0, sb1;

  // ---- prologue: fetch + stage k = 0..63 into buffer 0 ----
#pragma unroll
  for (int j = 0; j < 32; ++j) ra[j] = aptr[j];
  sa0 = sxa[0]; sa1 = sxa[1];
#pragma unroll
  for (int j = 0; j < 32; ++j) rb[j] = bptr[j];
  sb0 = swb[0]; sb1 = swb[1];
#pragma unroll
  for (int j = 0; j < 32; ++j) dA0[j] = (_Float16)(ra[j] * (j < 16 ? sa0 : sa1));
#pragma unroll
  for (int j = 0; j < 32; ++j) dB0[j] = (_Float16)(rb[j] * (j < 16 ? sb0 : sb1));
  __syncthreads();

  // ---- pipelined main loop: one barrier per 64-deep K-stage (16 WMMAs) ----
  for (int it = 0; it < NSTG - 1; ++it) {
    const int cur    = it & 1;
    const int nxtOff = (cur ^ 1) * bufStride;   // LDS offset of the write buffer
    const int k1     = (it + 1) * BK;

    // 1) issue next-stage global loads (must stay in flight under the WMMAs)
#pragma unroll
    for (int j = 0; j < 32; ++j) ra[j] = aptr[k1 + j];
    sa0 = sxa[(k1 >> 4) + 0]; sa1 = sxa[(k1 >> 4) + 1];
#pragma unroll
    for (int j = 0; j < 32; ++j) rb[j] = bptr[k1 + j];
    sb0 = swb[(k1 >> 4) + 0]; sb1 = swb[(k1 >> 4) + 1];

    // Pin the global loads ABOVE the compute block: without this the
    // scheduler sinks them next to their converts below the WMMAs and the
    // HBM latency is serialized instead of hidden under matrix work.
    __builtin_amdgcn_sched_barrier(0);

    // 2) compute on current buffer: 2 K-halves x (2 A-frags x 4 B-frags)
#pragma unroll
    for (int kh = 0; kh < 2; ++kh) {
      const int ko = kh * 32;   // half offset within the 64-deep stage
      v16h afr[2], bfr[4];
#pragma unroll
      for (int i = 0; i < 2; ++i)
        afr[i] = frag_from_lds(&As[cur][(mw * 32 + i * 16 + laneRow) * LDTK + ko], hiHalf);
#pragma unroll
      for (int j = 0; j < 4; ++j)
        bfr[j] = frag_from_lds(&Bs[cur][(nw * 64 + j * 16 + laneRow) * LDTK + ko], hiHalf);
#pragma unroll
      for (int i = 0; i < 2; ++i)
#pragma unroll
        for (int j = 0; j < 4; ++j)
          acc[i][j] = __builtin_amdgcn_wmma_f32_16x16x32_f16(
              false, afr[i], false, bfr[j], (short)0, acc[i][j], false, false);
    }

    // Keep the convert+store block below the WMMAs (the s_wait_loadcnt for
    // the staging registers then lands after the matrix work).
    __builtin_amdgcn_sched_barrier(0);

    // 3) convert + store next stage into the other buffer
    _Float16* const da = dA0 + nxtOff;
    _Float16* const db = dB0 + nxtOff;
#pragma unroll
    for (int j = 0; j < 32; ++j) da[j] = (_Float16)(ra[j] * (j < 16 ? sa0 : sa1));
#pragma unroll
    for (int j = 0; j < 32; ++j) db[j] = (_Float16)(rb[j] * (j < 16 ? sb0 : sb1));

    __syncthreads();
  }

  // ---- final K-stage: compute only ----
  {
    const int cur = (NSTG - 1) & 1;
#pragma unroll
    for (int kh = 0; kh < 2; ++kh) {
      const int ko = kh * 32;
      v16h afr[2], bfr[4];
#pragma unroll
      for (int i = 0; i < 2; ++i)
        afr[i] = frag_from_lds(&As[cur][(mw * 32 + i * 16 + laneRow) * LDTK + ko], hiHalf);
#pragma unroll
      for (int j = 0; j < 4; ++j)
        bfr[j] = frag_from_lds(&Bs[cur][(nw * 64 + j * 16 + laneRow) * LDTK + ko], hiHalf);
#pragma unroll
      for (int i = 0; i < 2; ++i)
#pragma unroll
        for (int j = 0; j < 4; ++j)
          acc[i][j] = __builtin_amdgcn_wmma_f32_16x16x32_f16(
              false, afr[i], false, bfr[j], (short)0, acc[i][j], false, false);
    }
  }

  // ---- epilogue: y = acc * gscale + bias (C/D layout: VGPR r -> M=r | r+8) ----
  const float g    = gscale[0];
  const int   mrow = (lane >> 4) << 3;
#pragma unroll
  for (int i = 0; i < 2; ++i) {
    const int mb = m0 + mw * 32 + i * 16 + mrow;
#pragma unroll
    for (int j = 0; j < 4; ++j) {
      const int n  = n0 + nw * 64 + j * 16 + laneRow;
      const int nc = min(n, Cout - 1);
      const float bv = bias[nc];
      if (n < Cout) {
#pragma unroll
        for (int r = 0; r < 8; ++r) {
          const int m = mb + r;
          if (m < T) out[(size_t)m * Cout + n] = acc[i][j][r] * g + bv;
        }
      }
    }
  }
}

extern "C" void kernel_launch(void* const* d_in, const int* in_sizes, int n_in,
                              void* d_out, int out_size, void* d_ws, size_t ws_size,
                              hipStream_t stream) {
  const float* qx   = (const float*)d_in[0];   // [T, 1024]
  const float* W    = (const float*)d_in[1];   // [1024, 1024]
  const float* sx   = (const float*)d_in[2];   // [T, 64]
  const float* sw   = (const float*)d_in[3];   // [1024, 64]
  const float* gs   = (const float*)d_in[4];   // [1]
  const float* bias = (const float*)d_in[5];   // [1024]
  float* out = (float*)d_out;

  const int Cout = in_sizes[5];                // 1024
  const int T    = in_sizes[0] / CIN;          // 65536

  dim3 grid((Cout + BN - 1) / BN, (T + BM - 1) / BM);   // 8 x 512
  nvfp4_qlinear_wmma<<<grid, 256, 0, stream>>>(qx, W, sx, sw, gs, bias, out, T, Cout);
}
